// GraphAttentionLayer_18734647345264
// MI455X (gfx1250) — compile-verified, matrix operand order checked by reading, not run on hardware
//
#include <hip/hip_runtime.h>
#include <hip/hip_bf16.h>
#include <math.h>

typedef __attribute__((ext_vector_type(16))) __bf16 v16bf;
typedef __attribute__((ext_vector_type(8)))  __bf16 v8bf;
typedef __attribute__((ext_vector_type(8)))  float  v8f;
typedef __attribute__((ext_vector_type(4)))  float  v4f;
typedef __attribute__((ext_vector_type(4)))  int    v4i;

#define D 128
#define NSEQ 2048
#define NB 8
#define NEG_BIG (-1.0e30f)

// Load a 16-half WMMA operand fragment: 8 halfs at p, 8 halfs at p+16.
// Caller supplies p = rowptr + kbase + (half16 ? 8 : 0); this matches the
// CDNA5 16-bit A/B layout (V0-3 = K0..7 | K8..15 ; V4-7 = K16..23 | K24..31).
__device__ __forceinline__ v16bf load_ab(const __bf16* __restrict__ p) {
  v8bf lo = *reinterpret_cast<const v8bf*>(p);
  v8bf hi = *reinterpret_cast<const v8bf*>(p + 16);
  return __builtin_shufflevector(lo, hi, 0,1,2,3,4,5,6,7,8,9,10,11,12,13,14,15);
}

// Exchange value between lane L and L^16 (SWAPX16 group-of-32 swizzle).
__device__ __forceinline__ float swap16(float x) {
  return __int_as_float(__builtin_amdgcn_ds_swizzle(__float_as_int(x), 0x401f));
}

// ---------------- Phase 0: layernorm + psi projection --------------------
// One block per row (b*n rows). Writes bf16 H copy and bf16 transposed
// messages Mt[b][d][n].
__global__ __launch_bounds__(128) void prep_kernel(
    const float* __restrict__ H, const float* __restrict__ gamma,
    const float* __restrict__ beta, const float* __restrict__ W,
    const float* __restrict__ bpsi, __bf16* __restrict__ Hb,
    __bf16* __restrict__ Mt) {
  const int row  = blockIdx.x;          // 0 .. NB*NSEQ-1
  const int b    = row >> 11;           // / NSEQ
  const int nidx = row & (NSEQ - 1);
  const int tid  = threadIdx.x;         // 0..127 (= d column)

  __shared__ float red[D];
  __shared__ float ln[D];

  float x = H[(size_t)row * D + tid];
  Hb[(size_t)row * D + tid] = (__bf16)x;

  // mean
  red[tid] = x; __syncthreads();
  #pragma unroll
  for (int s = 64; s > 0; s >>= 1) {
    if (tid < s) red[tid] += red[tid + s];
    __syncthreads();
  }
  float mu = red[0] * (1.0f / D);
  __syncthreads();
  // variance
  float dx = x - mu;
  red[tid] = dx * dx; __syncthreads();
  #pragma unroll
  for (int s = 64; s > 0; s >>= 1) {
    if (tid < s) red[tid] += red[tid + s];
    __syncthreads();
  }
  float var = red[0] * (1.0f / D);
  __syncthreads();

  ln[tid] = dx * rsqrtf(var + 1e-5f) * gamma[tid] + beta[tid];
  __syncthreads();

  // message column `tid`: dot(ln_row, W[tid,:]) + b, relu
  const float* __restrict__ w = W + (size_t)tid * D;
  float acc = bpsi[tid];
  #pragma unroll 8
  for (int k = 0; k < D; ++k) acc = fmaf(ln[k], w[k], acc);
  acc = fmaxf(acc, 0.0f);
  Mt[((size_t)b * D + tid) * NSEQ + nidx] = (__bf16)acc;
}

// ---------------- Phase 1: masked flash attention ------------------------
// One wave per 16-row i-tile. Computes S^T = H_m x H_i^T with WMMA so the
// exp'd probabilities are directly in B-operand layout for out^T = M^T x P.
__global__ __launch_bounds__(128) void attn_kernel(
    const __bf16* __restrict__ Hb, const __bf16* __restrict__ Mt,
    const int* __restrict__ Adj, float* __restrict__ out) {
  const int wave = threadIdx.x >> 5;
  const int lane = threadIdx.x & 31;
  const int l16  = lane & 15;
  const int half = lane >> 4;          // 0 or 1

  const int tile  = blockIdx.x * 4 + wave;   // 0..1023
  const int b     = tile >> 7;               // / (NSEQ/16)
  const int ibase = (tile & 127) << 4;

  const float rscale = 1.0f / (11.31370849898476f + 1e-5f); // 1/(sqrt(128)+1e-5)

  // B operand for the score GEMM: rows i (fixed for this wave)
  const __bf16* __restrict__ hB =
      Hb + ((size_t)b * NSEQ + ibase + l16) * D + half * 8;
  v16bf bH0 = load_ab(hB +  0);
  v16bf bH1 = load_ab(hB + 32);
  v16bf bH2 = load_ab(hB + 64);
  v16bf bH3 = load_ab(hB + 96);

  v8f acc[8];
  #pragma unroll
  for (int t = 0; t < 8; ++t) acc[t] = (v8f){};
  float m_run = NEG_BIG;
  float l_run = 0.0f;

  const __bf16* __restrict__ hmBase =
      Hb + (size_t)b * NSEQ * D + (size_t)l16 * D + half * 8;
  const __bf16* __restrict__ mtBase =
      Mt + (size_t)b * D * NSEQ + (size_t)l16 * NSEQ + half * 8;
  const int* __restrict__ aBase =
      Adj + ((size_t)b * NSEQ + ibase + l16) * NSEQ + half * 8;

  for (int mb = 0; mb < NSEQ; mb += 32) {
    // ---- scores: S^T tiles (m rows mb..mb+15 and mb+16..mb+31) ----
    v8f c0 = (v8f){}, c1 = (v8f){};
    const __bf16* __restrict__ hm = hmBase + (size_t)mb * D;
    {
      v16bf a0, a1;
      a0 = load_ab(hm +  0);            a1 = load_ab(hm + 16 * D +  0);
      c0 = __builtin_amdgcn_wmma_f32_16x16x32_bf16(false, a0, false, bH0, (short)0, c0, false, false);
      c1 = __builtin_amdgcn_wmma_f32_16x16x32_bf16(false, a1, false, bH0, (short)0, c1, false, false);
      a0 = load_ab(hm + 32);            a1 = load_ab(hm + 16 * D + 32);
      c0 = __builtin_amdgcn_wmma_f32_16x16x32_bf16(false, a0, false, bH1, (short)0, c0, false, false);
      c1 = __builtin_amdgcn_wmma_f32_16x16x32_bf16(false, a1, false, bH1, (short)0, c1, false, false);
      a0 = load_ab(hm + 64);            a1 = load_ab(hm + 16 * D + 64);
      c0 = __builtin_amdgcn_wmma_f32_16x16x32_bf16(false, a0, false, bH2, (short)0, c0, false, false);
      c1 = __builtin_amdgcn_wmma_f32_16x16x32_bf16(false, a1, false, bH2, (short)0, c1, false, false);
      a0 = load_ab(hm + 96);            a1 = load_ab(hm + 16 * D + 96);
      c0 = __builtin_amdgcn_wmma_f32_16x16x32_bf16(false, a0, false, bH3, (short)0, c0, false, false);
      c1 = __builtin_amdgcn_wmma_f32_16x16x32_bf16(false, a1, false, bH3, (short)0, c1, false, false);
    }

    // ---- adjacency mask (128MB single-use stream: NT b128 loads) ----
    const int* __restrict__ ap = aBase + mb;
    // prefetch the mask line this lane will need 8 iterations from now
    __builtin_prefetch(ap + 8 * 32, 0, 1);
    v4i a0a = __builtin_nontemporal_load(reinterpret_cast<const v4i*>(ap));
    v4i a0b = __builtin_nontemporal_load(reinterpret_cast<const v4i*>(ap + 4));
    v4i a1a = __builtin_nontemporal_load(reinterpret_cast<const v4i*>(ap + 16));
    v4i a1b = __builtin_nontemporal_load(reinterpret_cast<const v4i*>(ap + 20));
    int msk0[8] = {a0a[0], a0a[1], a0a[2], a0a[3], a0b[0], a0b[1], a0b[2], a0b[3]};
    int msk1[8] = {a1a[0], a1a[1], a1a[2], a1a[3], a1b[0], a1b[1], a1b[2], a1b[3]};

    float s0[8], s1[8];
    #pragma unroll
    for (int v = 0; v < 8; ++v) {
      s0[v] = msk0[v] ? c0[v] * rscale : NEG_BIG;
      s1[v] = msk1[v] ? c1[v] * rscale : NEG_BIG;
    }

    // ---- online softmax (lane pair L,L+16 shares row i) ----
    float mx = NEG_BIG;
    #pragma unroll
    for (int v = 0; v < 8; ++v) mx = fmaxf(mx, fmaxf(s0[v], s1[v]));
    mx = fmaxf(mx, swap16(mx));
    float mnew  = fmaxf(m_run, mx);
    float alpha = __expf(m_run - mnew);
    m_run = mnew;

    float p0[8], p1[8], ls = 0.0f;
    #pragma unroll
    for (int v = 0; v < 8; ++v) {
      p0[v] = __expf(s0[v] - mnew);
      p1[v] = __expf(s1[v] - mnew);
      ls += p0[v] + p1[v];
    }
    l_run = l_run * alpha + ls + swap16(ls);

    #pragma unroll
    for (int t = 0; t < 8; ++t)
      #pragma unroll
      for (int v = 0; v < 8; ++v) acc[t][v] *= alpha;

    // pack P into B-operand layout (tile0 -> halfs 0..7, tile1 -> 8..15)
    v16bf P;
    #pragma unroll
    for (int v = 0; v < 8; ++v) {
      P[v]     = (__bf16)p0[v];
      P[v + 8] = (__bf16)p1[v];
    }

    // ---- out^T += M^T_chunk x P ----
    const __bf16* __restrict__ mt = mtBase + mb;
    #pragma unroll
    for (int t = 0; t < 8; ++t) {
      v16bf am = load_ab(mt + (size_t)(t * 16) * NSEQ);
      acc[t] = __builtin_amdgcn_wmma_f32_16x16x32_bf16(false, am, false, P, (short)0, acc[t], false, false);
    }
  }

  // ---- normalize and store (write-once: NT stores) ----
  float invl = 1.0f / l_run;
  float* __restrict__ orow =
      out + ((size_t)b * NSEQ + ibase + l16) * D + half * 8;
  #pragma unroll
  for (int t = 0; t < 8; ++t) {
    v4f o0 = {acc[t][0] * invl, acc[t][1] * invl, acc[t][2] * invl, acc[t][3] * invl};
    v4f o1 = {acc[t][4] * invl, acc[t][5] * invl, acc[t][6] * invl, acc[t][7] * invl};
    __builtin_nontemporal_store(o0, reinterpret_cast<v4f*>(orow + t * 16));
    __builtin_nontemporal_store(o1, reinterpret_cast<v4f*>(orow + t * 16 + 4));
  }
}

extern "C" void kernel_launch(void* const* d_in, const int* in_sizes, int n_in,
                              void* d_out, int out_size, void* d_ws, size_t ws_size,
                              hipStream_t stream) {
  const float* H     = (const float*)d_in[0];
  const int*   A     = (const int*)  d_in[1];
  const float* gamma = (const float*)d_in[2];
  const float* beta  = (const float*)d_in[3];
  const float* W     = (const float*)d_in[4];
  const float* bpsi  = (const float*)d_in[5];
  float* out = (float*)d_out;

  const size_t elems = (size_t)NB * NSEQ * D;        // 2M elements
  __bf16* Hb = (__bf16*)d_ws;                        // 4 MB
  __bf16* Mt = (__bf16*)((char*)d_ws + elems * sizeof(__bf16)); // next 4 MB

  prep_kernel<<<NB * NSEQ, 128, 0, stream>>>(H, gamma, beta, W, bpsi, Hb, Mt);
  attn_kernel<<<(NB * NSEQ / 16) / 4, 128, 0, stream>>>(Hb, Mt, A, out);
}
